// QuantumMultiheadAttention_18940805775798
// MI455X (gfx1250) — matrix operand order repaired and checked by reading, not compiled, not run on hardware
//
#include <hip/hip_runtime.h>
#include <hip/hip_bf16.h>
#include <stdint.h>

// Problem constants (from reference)
constexpr int kB   = 2;
constexpr int kN   = 2048;
constexpr int kE   = 1024;
constexpr int kH   = 16;
constexpr int kDh  = 64;      // E / H
constexpr int kTop = 64;
constexpr int kM   = kB * kN; // 4096 rows for the projection GEMMs

typedef __attribute__((ext_vector_type(16))) __bf16 v16bf;
typedef __attribute__((ext_vector_type(8)))  float  v8f;

union Frag16 {
    uint4 u[2];
    v16bf v;
};

// Build a 16-element bf16 WMMA operand from two 16-byte chunks
__device__ __forceinline__ v16bf ld_frag16(const __bf16* p0, const __bf16* p1) {
    Frag16 f;
    f.u[0] = *(const uint4*)p0;
    f.u[1] = *(const uint4*)p1;
    return f.v;
}

// CDNA5 async copy: 16 bytes global -> LDS, tracked by ASYNCcnt
__device__ __forceinline__ void async_ld_b128(uint32_t lds_off, const void* gaddr) {
    asm volatile("global_load_async_to_lds_b128 %0, %1, off"
                 :: "v"(lds_off), "v"(gaddr) : "memory");
}
__device__ __forceinline__ void wait_async_le0() {
    asm volatile("s_wait_asynccnt 0x0" ::: "memory");
}
__device__ __forceinline__ void wait_async_le3() {
    asm volatile("s_wait_asynccnt 0x3" ::: "memory");
}
__device__ __forceinline__ uint32_t lds_off32(const void* p) {
    return (uint32_t)(uintptr_t)p;   // low 32 bits of LDS aperture addr = LDS offset
}

// ---------------------------------------------------------------------------
// Elementwise fp32 -> bf16 (one-time conversion of inputs / weights)
// ---------------------------------------------------------------------------
__global__ void __launch_bounds__(256)
cvt_f32_bf16_kernel(const float* __restrict__ src, __bf16* __restrict__ dst, int n) {
    int i = (blockIdx.x * 256 + threadIdx.x) * 4;
    if (i < n) {
        float4 x = *(const float4*)(src + i);
        dst[i + 0] = (__bf16)x.x;
        dst[i + 1] = (__bf16)x.y;
        dst[i + 2] = (__bf16)x.z;
        dst[i + 3] = (__bf16)x.w;
    }
}

// ---------------------------------------------------------------------------
// bf16 GEMM: Out[m,n] = sum_k X[m,k] * W[n,k] + bias[n]   (X @ W.T + b)
// Block = 256 threads = 8 waves. Block tile 64(M) x 128(N).
// Wave (mw = wid&3, nw = wid>>2) owns 16(M) x 64(N) = 4 WMMA tiles.
// Double-buffered LDS K-slabs staged with global_load_async_to_lds_b128:
// slab k+1's DMA overlaps slab k's WMMAs (wait asynccnt<=3: in-order done).
// ---------------------------------------------------------------------------
constexpr int kABytes = 64 * 32 * 2;    // 4 KB per A slab
constexpr int kBBytes = 128 * 32 * 2;   // 8 KB per B slab

template <bool WRITE_F32>
__global__ void __launch_bounds__(256)
gemm_bf16_kernel(const __bf16* __restrict__ X, const __bf16* __restrict__ W,
                 const float* __restrict__ bias,
                 __bf16* __restrict__ outBf, float* __restrict__ outF,
                 int M, int Ntot, int K) {
    __shared__ __bf16 sA[2][64 * 32];    // 2 x 4 KB
    __shared__ __bf16 sB[2][128 * 32];   // 2 x 8 KB

    const int tid  = threadIdx.x;
    const int lane = tid & 31;
    const int wid  = tid >> 5;
    const int nlo  = lane & 15;
    const int hi   = lane >> 4;
    const int mw   = wid & 3;         // 0..3 : which 16-row slab of the 64
    const int nw   = wid >> 2;        // 0..1 : which 64-col group of the 128

    const int mBlock = blockIdx.x * 64;
    const int nBlock = blockIdx.y * 128;

    // Async staging addresses (fixed roles per thread across the K loop)
    const int arow = tid >> 2, acol = (tid & 3) * 8;        // A: 64x32, 8 bf16/thread
    const int brow = tid >> 1, bcol = (tid & 1) * 16;       // B: 128x32, 16 bf16/thread
    const __bf16* gA = X + (size_t)(mBlock + arow) * K + acol;
    const __bf16* gB = W + (size_t)(nBlock + brow) * K + bcol;
    const uint32_t ldsA0 = lds_off32(&sA[0][arow * 32 + acol]);
    const uint32_t ldsB0 = lds_off32(&sB[0][brow * 32 + bcol]);

    v8f acc[4] = {};

    // prologue: stage slab 0 into buffer 0
    async_ld_b128(ldsA0,      gA);
    async_ld_b128(ldsB0,      gB);
    async_ld_b128(ldsB0 + 32, gB + 16);

    int cur = 0;
    for (int kk = 0; kk < K; kk += 32) {
        const int nxt = kk + 32;
        if (nxt < K) {
            // stage next slab into the other buffer while we compute this one
            const uint32_t a1 = ldsA0 + (cur ^ 1) * kABytes;
            const uint32_t b1 = ldsB0 + (cur ^ 1) * kBBytes;
            async_ld_b128(a1,      gA + nxt);
            async_ld_b128(b1,      gB + nxt);
            async_ld_b128(b1 + 32, gB + nxt + 16);
            __builtin_prefetch(gB + nxt + 32, 0, 1);
            wait_async_le3();   // in-order completion => current slab landed
        } else {
            wait_async_le0();
        }
        __syncthreads();

        // ---- A fragment (shared by this wave's 4 output tiles) ----
        const __bf16* ap = &sA[cur][(mw * 16 + nlo) * 32 + hi * 16];
        const v16bf a = ld_frag16(ap, ap + 8);

        // ---- preload all 4 B fragments, then 4 back-to-back WMMAs ----
        const __bf16* bbase = &sB[cur][(nw * 64 + nlo) * 32 + hi * 16];
        const v16bf bf0 = ld_frag16(bbase,            bbase + 8);
        const v16bf bf1 = ld_frag16(bbase + 16 * 32,  bbase + 16 * 32 + 8);
        const v16bf bf2 = ld_frag16(bbase + 32 * 32,  bbase + 32 * 32 + 8);
        const v16bf bf3 = ld_frag16(bbase + 48 * 32,  bbase + 48 * 32 + 8);
        acc[0] = __builtin_amdgcn_wmma_f32_16x16x32_bf16(false, a, false, bf0, (short)0, acc[0], false, false);
        acc[1] = __builtin_amdgcn_wmma_f32_16x16x32_bf16(false, a, false, bf1, (short)0, acc[1], false, false);
        acc[2] = __builtin_amdgcn_wmma_f32_16x16x32_bf16(false, a, false, bf2, (short)0, acc[2], false, false);
        acc[3] = __builtin_amdgcn_wmma_f32_16x16x32_bf16(false, a, false, bf3, (short)0, acc[3], false, false);

        __syncthreads();
        cur ^= 1;
    }

    // C layout: lanes 0-15 -> M = r, lanes 16-31 -> M = 8 + r; N = lane&15
    const int m0 = mBlock + mw * 16 + hi * 8;
#pragma unroll
    for (int nt = 0; nt < 4; ++nt) {
        const int n = nBlock + nw * 64 + nt * 16 + nlo;
        const float bb = bias[n];
#pragma unroll
        for (int r = 0; r < 8; ++r) {
            const float f = acc[nt][r] + bb;
            const size_t idx = (size_t)(m0 + r) * Ntot + n;
            if constexpr (WRITE_F32) outF[idx] = f;
            else                     outBf[idx] = (__bf16)f;
        }
    }
}

// ---------------------------------------------------------------------------
// Fused per-head scores (WMMA) + top-64 select + softmax + V gather.
// Grid: (N/16, B*H). Block = 256 threads = 8 waves. Q/K/V are bf16 planes.
// ---------------------------------------------------------------------------
__global__ void __launch_bounds__(256)
topk_attn_kernel(const __bf16* __restrict__ Qb, const __bf16* __restrict__ Kb,
                 const __bf16* __restrict__ Vb, __bf16* __restrict__ Attn) {
    __shared__ __bf16 s_scores[16 * kN];   // 16 rows x 2048 keys = 64 KB

    const int lane = threadIdx.x & 31;
    const int wid  = threadIdx.x >> 5;
    const int nlo  = lane & 15;
    const int hi   = lane >> 4;

    const int bh = blockIdx.y;
    const int b  = bh / kH;
    const int h  = bh % kH;
    const int q0 = blockIdx.x * 16;

    // ---- A: 16x64 query block (two 16x32 WMMA A operands) ----
    const __bf16* qbase = Qb + (size_t)(b * kN + q0 + nlo) * kE + h * kDh;
    const int k0 = hi * 16;
    const v16bf a0 = ld_frag16(qbase + k0,      qbase + k0 + 8);
    const v16bf a1 = ld_frag16(qbase + 32 + k0, qbase + 32 + k0 + 8);

    // ---- scores: each wave computes 16 key tiles, one-tile lookahead ----
    const __bf16* kcol = Kb + (size_t)(b * kN) * kE + h * kDh + hi * 16;
    auto keyptr = [&](int kt) { return kcol + (size_t)(kt * 16 + nlo) * kE; };

    v16bf b0 = ld_frag16(keyptr(wid * 16),      keyptr(wid * 16) + 8);
    v16bf b1 = ld_frag16(keyptr(wid * 16) + 32, keyptr(wid * 16) + 40);
    for (int t = 0; t < 16; ++t) {
        const int kt = wid * 16 + t;
        v16bf n0, n1;
        if (t < 15) {
            const __bf16* np = keyptr(kt + 1);
            n0 = ld_frag16(np,      np + 8);
            n1 = ld_frag16(np + 32, np + 40);
        }
        v8f c = {};
        c = __builtin_amdgcn_wmma_f32_16x16x32_bf16(false, a0, false, b0, (short)0, c, false, false);
        c = __builtin_amdgcn_wmma_f32_16x16x32_bf16(false, a1, false, b1, (short)0, c, false, false);
        const int m0 = hi * 8;
#pragma unroll
        for (int r = 0; r < 8; ++r)
            s_scores[(m0 + r) * kN + kt * 16 + nlo] = (__bf16)(c[r] * 0.125f); // 1/sqrt(64)
        b0 = n0;
        b1 = n1;
    }
    __syncthreads();

    // ---- per-row top-64 + softmax + gather; wave w owns rows 2w, 2w+1 ----
    for (int rr = 0; rr < 2; ++rr) {
        const int row  = wid * 2 + rr;
        const int qrow = q0 + row;
        __bf16* s = &s_scores[row * kN];

        float selv[2] = {-1e30f, -1e30f};
        int   seli[2] = {0, 0};

        for (int j = 0; j < kTop; ++j) {
            float bestv = -1e30f;
            int   besti = lane;
            for (int i = lane; i < kN; i += 32) {
                const float v = (float)s[i];
                if (v > bestv) { bestv = v; besti = i; }
            }
#pragma unroll
            for (int off = 16; off > 0; off >>= 1) {
                const float ov = __shfl_xor(bestv, off);
                const int   oi = __shfl_xor(besti, off);
                if (ov > bestv || (ov == bestv && oi < besti)) { bestv = ov; besti = oi; }
            }
            if (lane == (j & 31)) { selv[j >> 5] = bestv; seli[j >> 5] = besti; }
            if (lane == 0) s[besti] = (__bf16)(-1e30f);   // knock out the winner
        }

        // softmax over the 64 selected scores (2 per lane)
        float mx = fmaxf(selv[0], selv[1]);
#pragma unroll
        for (int off = 16; off > 0; off >>= 1) mx = fmaxf(mx, __shfl_xor(mx, off));
        const float e0 = __expf(selv[0] - mx);
        const float e1 = __expf(selv[1] - mx);
        float ssum = e0 + e1;
#pragma unroll
        for (int off = 16; off > 0; off >>= 1) ssum += __shfl_xor(ssum, off);
        const float w0 = e0 / ssum;
        const float w1 = e1 / ssum;

        // gather V rows: lane owns dims {2*lane, 2*lane+1}
        float acc0 = 0.f, acc1 = 0.f;
        const int d0 = 2 * lane;
        for (int j = 0; j < kTop; ++j) {
            const int src  = j & 31;
            const int slot = j >> 5;
            const int   kidx = __shfl(slot ? seli[1] : seli[0], src);
            const float wv   = __shfl(slot ? w1 : w0, src);
            const __bf16* vp = Vb + (size_t)(b * kN + kidx) * kE + h * kDh + d0;
            acc0 += wv * (float)vp[0];
            acc1 += wv * (float)vp[1];
        }
        __bf16* op = Attn + (size_t)(b * kN + qrow) * kE + h * kDh + d0;
        op[0] = (__bf16)acc0;
        op[1] = (__bf16)acc1;
    }
}

// ---------------------------------------------------------------------------
extern "C" void kernel_launch(void* const* d_in, const int* in_sizes, int n_in,
                              void* d_out, int out_size, void* d_ws, size_t ws_size,
                              hipStream_t stream) {
    const float* query = (const float*)d_in[0];
    const float* key   = (const float*)d_in[1];
    const float* value = (const float*)d_in[2];
    const float* Wq    = (const float*)d_in[3];
    const float* bq    = (const float*)d_in[4];
    const float* Wk    = (const float*)d_in[5];
    const float* bk    = (const float*)d_in[6];
    const float* Wv    = (const float*)d_in[7];
    const float* bv    = (const float*)d_in[8];
    const float* Wo    = (const float*)d_in[9];
    const float* bo    = (const float*)d_in[10];
    float* out = (float*)d_out;

    const size_t actElems = (size_t)kM * kE;   // 4M
    const size_t wElems   = (size_t)kE * kE;   // 1M

    __bf16* xq  = (__bf16*)d_ws;               // bf16 input planes
    __bf16* xk  = xq  + actElems;
    __bf16* xv  = xk  + actElems;
    __bf16* wqb = xv  + actElems;              // bf16 weights
    __bf16* wkb = wqb + wElems;
    __bf16* wvb = wkb + wElems;
    __bf16* wob = wvb + wElems;
    __bf16* Qb  = wob + wElems;                // projected planes (bf16)
    __bf16* Kb2 = Qb  + actElems;
    __bf16* Vb2 = Kb2 + actElems;
    __bf16* Atb = Vb2 + actElems;              // attention output (bf16)
    // total: 32M bf16 = 64 MB of workspace

    dim3 bb(256);

    // 1) one-time fp32 -> bf16 conversions
    cvt_f32_bf16_kernel<<<dim3(actElems / (256 * 4)), bb, 0, stream>>>(query, xq, (int)actElems);
    cvt_f32_bf16_kernel<<<dim3(actElems / (256 * 4)), bb, 0, stream>>>(key,   xk, (int)actElems);
    cvt_f32_bf16_kernel<<<dim3(actElems / (256 * 4)), bb, 0, stream>>>(value, xv, (int)actElems);
    cvt_f32_bf16_kernel<<<dim3(wElems   / (256 * 4)), bb, 0, stream>>>(Wq, wqb, (int)wElems);
    cvt_f32_bf16_kernel<<<dim3(wElems   / (256 * 4)), bb, 0, stream>>>(Wk, wkb, (int)wElems);
    cvt_f32_bf16_kernel<<<dim3(wElems   / (256 * 4)), bb, 0, stream>>>(Wv, wvb, (int)wElems);
    cvt_f32_bf16_kernel<<<dim3(wElems   / (256 * 4)), bb, 0, stream>>>(Wo, wob, (int)wElems);

    // 2) Q/K/V projections (bf16 out)
    dim3 gg(kM / 64, kE / 128);   // 64 x 8 blocks
    gemm_bf16_kernel<false><<<gg, bb, 0, stream>>>(xq, wqb, bq, Qb,  nullptr, kM, kE, kE);
    gemm_bf16_kernel<false><<<gg, bb, 0, stream>>>(xk, wkb, bk, Kb2, nullptr, kM, kE, kE);
    gemm_bf16_kernel<false><<<gg, bb, 0, stream>>>(xv, wvb, bv, Vb2, nullptr, kM, kE, kE);

    // 3) fused top-k attention (bf16 out)
    dim3 ga(kN / 16, kB * kH);    // 128 x 32 blocks
    topk_attn_kernel<<<ga, bb, 0, stream>>>(Qb, Kb2, Vb2, Atb);

    // 4) output projection (fp32 out -> d_out)
    gemm_bf16_kernel<true><<<gg, bb, 0, stream>>>(Atb, wob, bo, nullptr, out, kM, kE, kE);
}
